// QINSLinear_14293651161454
// MI455X (gfx1250) — matrix operand order repaired and checked by reading, not compiled, not run on hardware
//
#include <hip/hip_runtime.h>
#include <cstdint>

typedef _Float16 v16h  __attribute__((ext_vector_type(16)));
typedef __fp16   f16x2 __attribute__((ext_vector_type(2)));   // cvt_pkrtz return type
typedef float    v8f   __attribute__((ext_vector_type(8)));

#define IN_F  4096
#define OUT_F 11008
#define BATCH 8

union AB16 { v16h v; f16x2 p[8]; uint32_t u[8]; uint4 q[2]; };

static __device__ __forceinline__ uint32_t f2u(f16x2 p) {
    union { f16x2 p; uint32_t u; } c; c.p = p; return c.u;
}

// ---- pre-kernel: pack x (8 x 4096 f32) to f16 row-major into workspace ----
__global__ __launch_bounds__(256)
void pack_x_f16(const float* __restrict__ x, _Float16* __restrict__ xh)
{
    const int t = (blockIdx.x * 256 + threadIdx.x) * 4;   // 8192 threads x 4 elems
    float4 v = *(const float4*)(x + t);
    union { f16x2 p[2]; uint2 u; } r;
    r.p[0] = __builtin_amdgcn_cvt_pkrtz(v.x, v.y);
    r.p[1] = __builtin_amdgcn_cvt_pkrtz(v.z, v.w);
    *(uint2*)(xh + t) = r.u;
}

// Block = 256 threads = 8 wave32. Block owns 16 output columns.
// Wave w owns K-slice [w*512, w*512+512) -> 16 x v_wmma_f32_16x16x32_f16.
// Cross-wave partial sums reduced through LDS, bias added, rows 0..7 stored.
template<bool PRE>
__global__ __launch_bounds__(256)
void qins_linear_wmma(const float* __restrict__ x,
                      const _Float16* __restrict__ xh,
                      const uint8_t* __restrict__ stored,
                      const int8_t* __restrict__ sign,
                      const float* __restrict__ log_min,
                      const float* __restrict__ log_max,
                      const float* __restrict__ bias,
                      float* __restrict__ out)
{
    __shared__ float red[8 * 256];   // 8 KB: per-wave 16x16 f32 partial tiles

    const int tid  = threadIdx.x;
    const int wave = tid >> 5;
    const int lane = tid & 31;
    const int half = lane >> 4;        // 0: lanes 0-15, 1: lanes 16-31
    const int nloc = lane & 15;        // output column within tile
    const int n    = blockIdx.x * 16 + nloc;
    const int mrow = nloc & 7;         // clamped A row: rows 8-15 mirror 0-7 (never stored)

    // decode constants: mag = exp2(d0 + code*d1); sign via bit 7 of sign byte
    const float lmin  = log_min[0];
    const float range = log_max[0] - lmin;
    const float LOG2E = 1.44269504088896340736f;
    const float d0 =  LOG2E * (lmin + range * (255.0f / 254.0f));
    const float d1 = -LOG2E * (range / 254.0f);

    const int kw = wave * 512;

    v8f c = {};

    #pragma unroll 2
    for (int i = 0; i < 16; ++i) {
        const int k0 = kw + i * 32;

        // ---- A fragment: rows of x, two runs of 8 values per lane ----
        // lanes 0-15: K = k0+0..15 ; lanes 16-31: K = k0+16..31
        AB16 A;
        if constexpr (PRE) {
            const _Float16* xr = xh + (size_t)mrow * IN_F;
            A.q[0] = *(const uint4*)(xr + k0 + 16 * half);        // 8 f16 = 16B
            A.q[1] = *(const uint4*)(xr + k0 + 16 * half + 8);
        } else {
            const float* xr = x + (size_t)mrow * IN_F;
            const float4* xa = (const float4*)(xr + k0 + 16 * half);
            const float4* xb = (const float4*)(xr + k0 + 16 * half + 8);
            float4 a0 = xa[0], a1 = xa[1];
            float4 b0 = xb[0], b1 = xb[1];
            A.p[0] = __builtin_amdgcn_cvt_pkrtz(a0.x, a0.y);
            A.p[1] = __builtin_amdgcn_cvt_pkrtz(a0.z, a0.w);
            A.p[2] = __builtin_amdgcn_cvt_pkrtz(a1.x, a1.y);
            A.p[3] = __builtin_amdgcn_cvt_pkrtz(a1.z, a1.w);
            A.p[4] = __builtin_amdgcn_cvt_pkrtz(b0.x, b0.y);
            A.p[5] = __builtin_amdgcn_cvt_pkrtz(b0.z, b0.w);
            A.p[6] = __builtin_amdgcn_cvt_pkrtz(b1.x, b1.y);
            A.p[7] = __builtin_amdgcn_cvt_pkrtz(b1.z, b1.w);
        }

        // ---- B fragment: 16 contiguous codes + signs per lane, one b128 each ----
        // lanes 0-15 cover K = k0..k0+15 of column n; lanes 16-31 cover k0+16..k0+31
        const size_t woff = (size_t)n * IN_F + (size_t)(k0 + 16 * half);
        uint4 q  = *(const uint4*)(stored + woff);
        uint4 sg = *(const uint4*)((const uint8_t*)sign + woff);
        const uint32_t qs[4] = { q.x, q.y, q.z, q.w };
        const uint32_t ss[4] = { sg.x, sg.y, sg.z, sg.w };
        AB16 B;
        #pragma unroll
        for (int d = 0; d < 4; ++d) {
            const uint32_t qq = qs[d];
            float m0 = __builtin_amdgcn_exp2f(fmaf((float)( qq        & 0xFFu), d1, d0));
            float m1 = __builtin_amdgcn_exp2f(fmaf((float)((qq >>  8) & 0xFFu), d1, d0));
            float m2 = __builtin_amdgcn_exp2f(fmaf((float)((qq >> 16) & 0xFFu), d1, d0));
            float m3 = __builtin_amdgcn_exp2f(fmaf((float)( qq >> 24        ), d1, d0));
            // packed sign masks via v_perm_b32: pre-mask sign bytes to 0x80, then
            // place byte0 at bits[15:8] (f16-lo sign) and byte1 at bits[31:24]
            // (f16-hi sign); selector 0x0C injects 0x00 bytes. Result needs no
            // further masking: {sdm.b1, 0, sdm.b0, 0}.
            const uint32_t sdm = ss[d] & 0x80808080u;
            uint32_t s01 = __builtin_amdgcn_perm(0u, sdm, 0x010C000Cu);
            uint32_t s23 = __builtin_amdgcn_perm(0u, sdm, 0x030C020Cu);
            B.u[2 * d]     = f2u(__builtin_amdgcn_cvt_pkrtz(m0, m1)) | s01;
            B.u[2 * d + 1] = f2u(__builtin_amdgcn_cvt_pkrtz(m2, m3)) | s23;
        }

        // D = A x B + C   (16x16x32 f16 -> f32)
        c = __builtin_amdgcn_wmma_f32_16x16x32_f16(false, A.v, false, B.v,
                                                   (short)0, c, false, false);
    }

    // ---- cross-wave K reduction through LDS ----
    // lane's 8 C values stored contiguously: position p = lane*8 + r in wave's block
    float* myred = &red[wave * 256 + lane * 8];
    ((float4*)myred)[0] = make_float4(c[0], c[1], c[2], c[3]);
    ((float4*)myred)[1] = make_float4(c[4], c[5], c[6], c[7]);
    __syncthreads();

    // thread t reduces position t across the 8 wave blocks.
    // p = l*8 + r, source lane l = t>>3, VGPR r = t&7 -> M = r + 8*(l>=16), N = l&15
    const int l = tid >> 3;
    const int r = tid & 7;
    float sum = 0.0f;
    #pragma unroll
    for (int w2 = 0; w2 < 8; ++w2) sum += red[w2 * 256 + tid];
    if (l < 16) {   // M = r in 0..7 (valid batch rows); l>=16 are clamped duplicates
        const int col = blockIdx.x * 16 + l;
        out[(size_t)r * OUT_F + col] = sum + bias[col];
    }
}

extern "C" void kernel_launch(void* const* d_in, const int* in_sizes, int n_in,
                              void* d_out, int out_size, void* d_ws, size_t ws_size,
                              hipStream_t stream) {
    const float*   x      = (const float*)d_in[0];
    const uint8_t* stored = (const uint8_t*)d_in[1];
    const int8_t*  sign   = (const int8_t*)d_in[2];
    const float*   lmin   = (const float*)d_in[3];
    const float*   lmax   = (const float*)d_in[4];
    const float*   bias   = (const float*)d_in[5];
    float* out = (float*)d_out;

    dim3 grid(OUT_F / 16);   // 688 blocks, one 16-column tile each
    dim3 block(256);         // 8 wave32 -> K split 8 x 512

    const size_t xh_bytes = (size_t)BATCH * IN_F * sizeof(_Float16);  // 64 KB
    if (ws_size >= xh_bytes) {
        _Float16* xh = (_Float16*)d_ws;
        pack_x_f16<<<dim3((BATCH * IN_F) / (256 * 4)), dim3(256), 0, stream>>>(x, xh);
        qins_linear_wmma<true><<<grid, block, 0, stream>>>(x, xh, stored, sign,
                                                           lmin, lmax, bias, out);
    } else {
        qins_linear_wmma<false><<<grid, block, 0, stream>>>(x, nullptr, stored, sign,
                                                            lmin, lmax, bias, out);
    }
}
